// SequenceTower_326417514541
// MI455X (gfx1250) — matrix operand order repaired and checked by reading, not compiled, not run on hardware
//
#include <hip/hip_runtime.h>
#include <hip/hip_bf16.h>

typedef __attribute__((ext_vector_type(16))) _Float16 v16h;
typedef __attribute__((ext_vector_type(8)))  _Float16 v8h;
typedef __attribute__((ext_vector_type(8)))  float    v8f;
typedef __attribute__((ext_vector_type(4)))  unsigned int u32x4;
typedef __attribute__((ext_vector_type(8)))  int i32x8;
typedef __attribute__((ext_vector_type(4)))  int i32x4;

#define L_SEQ   8192
#define HDIM    1024
#define NLAYERS 6
#define CHUNK   256
#define NCHUNK  (L_SEQ / CHUNK)

#if defined(__has_builtin)
#if __has_builtin(__builtin_amdgcn_tensor_load_to_lds)
#define HAVE_TDM 1
#endif
#endif
#ifndef HAVE_TDM
#define HAVE_TDM 0
#endif

__device__ __forceinline__ float sigm(float x) { return 1.0f / (1.0f + __expf(-x)); }

// ---------------- fp32 -> fp16 convert (row-major copy) ----------------
__global__ __launch_bounds__(256) void k_f32_to_f16(const float* __restrict__ s,
                                                    _Float16* __restrict__ d, int n) {
  int i = blockIdx.x * 256 + threadIdx.x;
  int st = gridDim.x * 256;
  for (; i < n; i += st) d[i] = (_Float16)s[i];
}

// ---------------- fp32[K,N] -> fp16[N,K] transposed convert (per layer z) ----------------
__global__ __launch_bounds__(256) void k_transconv(const float* __restrict__ src,
                                                   _Float16* __restrict__ dst, int K, int N) {
  __shared__ _Float16 t[32][33];
  const int l = blockIdx.z;
  const float* s = src + (size_t)l * K * N;
  _Float16* d = dst + (size_t)l * N * K;
  const int k0 = blockIdx.y * 32, n0 = blockIdx.x * 32;
  const int tx = threadIdx.x, ty = threadIdx.y;
  for (int r = ty; r < 32; r += 8)
    t[r][tx] = (_Float16)s[(size_t)(k0 + r) * N + n0 + tx];
  __syncthreads();
  for (int r = ty; r < 32; r += 8)
    d[(size_t)(n0 + r) * K + k0 + tx] = t[tx][r];
}

// ---------------- LayerNorm (one row per block) ----------------
template <bool OUT16>
__global__ __launch_bounds__(256) void k_layernorm(const float* __restrict__ x,
                                                   const float* __restrict__ g,
                                                   const float* __restrict__ b,
                                                   void* __restrict__ outp) {
  __shared__ float s1[256], s2[256];
  const int row = blockIdx.x, tid = threadIdx.x;
  const float* xr = x + (size_t)row * HDIM;
  float a = 0.f, q = 0.f;
  for (int i = tid; i < HDIM; i += 256) { float v = xr[i]; a += v; q += v * v; }
  s1[tid] = a; s2[tid] = q; __syncthreads();
  for (int o = 128; o > 0; o >>= 1) {
    if (tid < o) { s1[tid] += s1[tid + o]; s2[tid] += s2[tid + o]; }
    __syncthreads();
  }
  const float mean = s1[0] * (1.f / HDIM);
  const float var  = s2[0] * (1.f / HDIM) - mean * mean;
  const float inv  = rsqrtf(var + 1e-5f);
  for (int i = tid; i < HDIM; i += 256) {
    float v = (xr[i] - mean) * inv * g[i] + b[i];
    if (OUT16) ((_Float16*)outp)[(size_t)row * HDIM + i] = (_Float16)v;
    else       ((float*)outp)[(size_t)row * HDIM + i] = v;
  }
}

// ---------------- TDM tile load: 128 rows x 32 halves -> LDS (padded to LDT=40) ----------------
#define BM  128
#define BN  128
#define BKK 32
#define LDT 40  // 32 data halves + 8 pad halves per row (pad added by TDM)

#if HAVE_TDM
__device__ __forceinline__ void tdm_load_tile_128x32(const _Float16* gsrc, unsigned ldsOff,
                                                     int rowStrideElems, int tensorRows) {
  const unsigned long long ga = (unsigned long long)(const void*)gsrc;
  u32x4 g0;
  g0[0] = 1u;                                   // count=1, user descriptor
  g0[1] = ldsOff;                               // LDS byte address
  g0[2] = (unsigned)ga;                         // global_addr[31:0]
  g0[3] = (unsigned)((ga >> 32) & 0x01FFFFFFull) | (2u << 30);  // ga[56:32] | type=2
  i32x8 g1;
  // data_size=2B(1<<16) | pad_enable(1<<20) | pad_interval=16dw(3<<22) | pad_amount=4dw(3<<25)
  g1[0] = (int)((1u << 16) | (1u << 20) | (3u << 22) | (3u << 25));
  g1[1] = (int)((unsigned)(rowStrideElems & 0xFFFF) << 16);  // tensor_dim0[15:0] @bits63:48
  g1[2] = (int)((unsigned)(tensorRows & 0xFFFF) << 16);      // tensor_dim1[15:0] @bits95:80
  g1[3] = (int)(32u << 16);                                  // tile_dim0=32 @bits127:112
  g1[4] = 128;                                               // tile_dim1=128 @bits143:128
  g1[5] = rowStrideElems;                                    // tensor_dim0_stride[31:0]
  g1[6] = 0; g1[7] = 0;
  i32x4 z = {0, 0, 0, 0};
#if __has_include(<hip/amd_detail/amd_gfx1250_TDM.h>)
  i32x8 z8 = {0, 0, 0, 0, 0, 0, 0, 0};
  __builtin_amdgcn_tensor_load_to_lds(g0, g1, z, z, z8, 0);
#else
  __builtin_amdgcn_tensor_load_to_lds(g0, g1, z, z, 0);
#endif
}
#endif

// ---------------- WMMA GEMM: C = A(f16)[M,K] @ Bt(f16)[N,K]^T (+bias, epilogue) ----------------
// EPI: 0 = store f32, 1 = sigmoid->f32, 2 = silu->f16, 3 = residual add into f32
template <int EPI>
__global__ __launch_bounds__(256, 2) void k_gemm_f16(
    const _Float16* __restrict__ A,   // [M,K] row-major
    const _Float16* __restrict__ Bt,  // [N,K] row-major (weights pre-transposed)
    const float* __restrict__ bias, float* __restrict__ outF,
    _Float16* __restrict__ outH, int M, int N, int K) {
  __shared__ alignas(128) _Float16 As[2][BM * LDT];  // [m][k]
  __shared__ alignas(128) _Float16 Bs[2][BN * LDT];  // [n][k]
  const int tid  = threadIdx.x;
  const int lane = tid & 31;
  const int wid  = tid >> 5;
  const int bm = blockIdx.y * BM;
  const int bn = blockIdx.x * BN;
  const int wm = (wid & 3) * 32;   // 4 waves along M
  const int wn = (wid >> 2) * 64;  // 2 waves along N

  // CDNA5 wave32 fragment lane decode (ISA 7.12.2)
  const int mA    = lane & 15;
  const int kAoff = (lane >> 4) * 8;
  const int nB    = lane & 15;
  const int kBoff = (lane >> 4) * 8;

  v8f acc[2][4] = {};
  const int nIter = K / BKK;

  auto mma_step = [&](const _Float16* AsP, const _Float16* BsP) {
    v16h af[2]; v16h bf[4];
#pragma unroll
    for (int ms = 0; ms < 2; ++ms) {
      const _Float16* p = AsP + (wm + ms * 16 + mA) * LDT + kAoff;
      const v8h lo = *(const v8h*)(p);       // K = kAoff .. kAoff+7
      const v8h hi = *(const v8h*)(p + 16);  // K = 16+kAoff .. +7
#pragma unroll
      for (int j = 0; j < 8; ++j) { af[ms][j] = lo[j]; af[ms][8 + j] = hi[j]; }
    }
#pragma unroll
    for (int ns = 0; ns < 4; ++ns) {
      const _Float16* p = BsP + (wn + ns * 16 + nB) * LDT + kBoff;
      const v8h lo = *(const v8h*)(p);
      const v8h hi = *(const v8h*)(p + 16);
#pragma unroll
      for (int j = 0; j < 8; ++j) { bf[ns][j] = lo[j]; bf[ns][8 + j] = hi[j]; }
    }
#pragma unroll
    for (int ms = 0; ms < 2; ++ms)
#pragma unroll
      for (int ns = 0; ns < 4; ++ns)
        acc[ms][ns] = __builtin_amdgcn_wmma_f32_16x16x32_f16(
            false, af[ms], false, bf[ns], (short)0, acc[ms][ns], false, false);
  };

#if HAVE_TDM
  // Double-buffered TDM pipeline: DMA stage it+1 overlaps WMMA on stage it.
  const unsigned asOff = (unsigned)(uintptr_t)(const void*)&As[0][0];
  const unsigned bsOff = (unsigned)(uintptr_t)(const void*)&Bs[0][0];
  const unsigned bufBytes = (unsigned)(BM * LDT * sizeof(_Float16));
  if (wid == 0) {
    tdm_load_tile_128x32(A + (size_t)bm * K, asOff, K, M);
    tdm_load_tile_128x32(Bt + (size_t)bn * K, bsOff, K, N);
  }
  for (int it = 0; it < nIter; ++it) {
    if (wid == 0) __builtin_amdgcn_s_wait_tensorcnt(0);  // stage `it` landed in LDS
    __syncthreads();  // all waves see stage it; all done reading buf[(it+1)&1]
    if (wid == 0 && it + 1 < nIter) {
      const int k0 = (it + 1) * BKK;
      const unsigned b = (unsigned)((it + 1) & 1);
      tdm_load_tile_128x32(A + (size_t)bm * K + k0, asOff + b * bufBytes, K, M);
      tdm_load_tile_128x32(Bt + (size_t)bn * K + k0, bsOff + b * bufBytes, K, N);
    }
    mma_step(As[it & 1], Bs[it & 1]);
  }
#else
  // Fallback: vector-load staging (both tiles are natural [row][32] slabs now)
  const int row = tid >> 1, half = tid & 1;
  for (int it = 0; it < nIter; ++it) {
    const int k0 = it * BKK;
    {
      const _Float16* sA = A + (size_t)(bm + row) * K + k0 + half * 16;
      const uint4 a0 = ((const uint4*)sA)[0];
      const uint4 a1 = ((const uint4*)sA)[1];
      *(uint4*)&As[0][row * LDT + half * 16]     = a0;
      *(uint4*)&As[0][row * LDT + half * 16 + 8] = a1;
      const _Float16* sB = Bt + (size_t)(bn + row) * K + k0 + half * 16;
      const uint4 b0 = ((const uint4*)sB)[0];
      const uint4 b1 = ((const uint4*)sB)[1];
      *(uint4*)&Bs[0][row * LDT + half * 16]     = b0;
      *(uint4*)&Bs[0][row * LDT + half * 16 + 8] = b1;
    }
    __syncthreads();
    mma_step(As[0], Bs[0]);
    __syncthreads();
  }
#endif

  // Epilogue. C/D layout: VGPR r -> M = r (lanes 0-15) / 8+r (lanes 16-31), N = lane&15.
#pragma unroll
  for (int ms = 0; ms < 2; ++ms)
#pragma unroll
    for (int ns = 0; ns < 4; ++ns) {
      const int mBase = bm + wm + ms * 16 + (lane >> 4) * 8;
      const int n = bn + wn + ns * 16 + (lane & 15);
      const float bv = bias[n];
#pragma unroll
      for (int r = 0; r < 8; ++r) {
        const float v = acc[ms][ns][r] + bv;
        const size_t o = (size_t)(mBase + r) * N + n;
        if constexpr (EPI == 0) outF[o] = v;
        else if constexpr (EPI == 1) outF[o] = sigm(v);
        else if constexpr (EPI == 2) outH[o] = (_Float16)(v * sigm(v));
        else outF[o] += v;
      }
    }
}

// ---------------- Chunked bidirectional scan ----------------
__global__ __launch_bounds__(256) void k_scan1(const float* __restrict__ u,
                                               const float* __restrict__ decay_p,
                                               float* __restrict__ fs, float* __restrict__ bs,
                                               float* __restrict__ cF, float* __restrict__ cB) {
  const int idx = blockIdx.x * 256 + threadIdx.x;  // H*NCHUNK threads
  const int c = idx & (HDIM - 1);
  const int chunk = idx >> 10;
  const float d = sigm(decay_p[c]);
  const int t0 = chunk * CHUNK;
  float s = 0.f;
  for (int i = 0; i < CHUNK; ++i) {
    s = d * s + u[(size_t)(t0 + i) * HDIM + c];
    fs[(size_t)(t0 + i) * HDIM + c] = s;
  }
  cF[chunk * HDIM + c] = s;
  s = 0.f;
  for (int i = CHUNK - 1; i >= 0; --i) {
    s = d * s + u[(size_t)(t0 + i) * HDIM + c];
    bs[(size_t)(t0 + i) * HDIM + c] = s;
  }
  cB[chunk * HDIM + c] = s;
}

__global__ __launch_bounds__(256) void k_scan2(float* __restrict__ cF, float* __restrict__ cB,
                                               const float* __restrict__ decay_p) {
  const int c = blockIdx.x * 256 + threadIdx.x;  // HDIM threads
  const float d = sigm(decay_p[c]);
  float dC = d;
  for (int i = 0; i < 8; ++i) dC *= dC;  // d^256
  float r = 0.f;
  for (int j = 0; j < NCHUNK; ++j) { r = dC * r + cF[j * HDIM + c]; cF[j * HDIM + c] = r; }
  r = 0.f;
  for (int j = NCHUNK - 1; j >= 0; --j) { r = dC * r + cB[j * HDIM + c]; cB[j * HDIM + c] = r; }
}

__global__ __launch_bounds__(256) void k_scan3(float* __restrict__ fs, const float* __restrict__ bs,
                                               const float* __restrict__ gate,
                                               const float* __restrict__ decay_p,
                                               const float* __restrict__ cF,
                                               const float* __restrict__ cB,
                                               _Float16* __restrict__ sout) {
  const int idx = blockIdx.x * 256 + threadIdx.x;
  const int c = idx & (HDIM - 1);
  const int chunk = idx >> 10;
  const float d = sigm(decay_p[c]);
  const float rF = (chunk > 0) ? cF[(chunk - 1) * HDIM + c] : 0.f;
  const float rB = (chunk < NCHUNK - 1) ? cB[(chunk + 1) * HDIM + c] : 0.f;
  const int t0 = chunk * CHUNK;
  float p = d;
  for (int i = 0; i < CHUNK; ++i) {
    const size_t o = (size_t)(t0 + i) * HDIM + c;
    fs[o] += p * rF;
    p *= d;
  }
  float pb = d;
  for (int i = CHUNK - 1; i >= 0; --i) {
    const size_t o = (size_t)(t0 + i) * HDIM + c;
    const float bsv = bs[o] + pb * rB;
    pb *= d;
    sout[o] = (_Float16)(0.5f * (fs[o] + bsv) * gate[o]);
  }
}

// ---------------- driver ----------------
extern "C" void kernel_launch(void* const* d_in, const int* in_sizes, int n_in,
                              void* d_out, int out_size, void* d_ws, size_t ws_size,
                              hipStream_t stream) {
  (void)in_sizes; (void)n_in; (void)out_size; (void)ws_size;
  const float* nf    = (const float*)d_in[0];
  const float* Wp    = (const float*)d_in[1];
  const float* bp    = (const float*)d_in[2];
  const float* ln1_g = (const float*)d_in[3];
  const float* ln1_b = (const float*)d_in[4];
  const float* Wi    = (const float*)d_in[5];
  const float* bi    = (const float*)d_in[6];
  const float* Wg    = (const float*)d_in[7];
  const float* bg    = (const float*)d_in[8];
  const float* Wo    = (const float*)d_in[9];
  const float* bo    = (const float*)d_in[10];
  const float* decay = (const float*)d_in[11];
  const float* ln2_g = (const float*)d_in[12];
  const float* ln2_b = (const float*)d_in[13];
  const float* W1    = (const float*)d_in[14];
  const float* b1    = (const float*)d_in[15];
  const float* W2    = (const float*)d_in[16];
  const float* b2    = (const float*)d_in[17];
  const float* lnf_g = (const float*)d_in[18];
  const float* lnf_b = (const float*)d_in[19];

  char* ws = (char*)d_ws;
  size_t off = 0;
  auto carve = [&](size_t bytes) -> void* {
    void* p = (void*)(ws + off);
    off = (off + bytes + 255) & ~(size_t)255;
    return p;
  };
  const size_t LH = (size_t)L_SEQ * HDIM;
  _Float16* nf16 = (_Float16*)carve(sizeof(_Float16) * (size_t)L_SEQ * 256);
  _Float16* wp16 = (_Float16*)carve(sizeof(_Float16) * (size_t)256 * HDIM);               // [N=H][K=256]
  _Float16* wi16 = (_Float16*)carve(sizeof(_Float16) * (size_t)NLAYERS * HDIM * HDIM);    // [N][K]
  _Float16* wg16 = (_Float16*)carve(sizeof(_Float16) * (size_t)NLAYERS * HDIM * HDIM);
  _Float16* wo16 = (_Float16*)carve(sizeof(_Float16) * (size_t)NLAYERS * HDIM * HDIM);
  _Float16* w116 = (_Float16*)carve(sizeof(_Float16) * (size_t)NLAYERS * HDIM * 2 * HDIM);  // [2H][H]
  _Float16* w216 = (_Float16*)carve(sizeof(_Float16) * (size_t)NLAYERS * HDIM * 2 * HDIM);  // [H][2H]
  float*    x32  = (float*)carve(sizeof(float) * LH);
  _Float16* h16  = (_Float16*)carve(sizeof(_Float16) * LH);
  float*    u32  = (float*)carve(sizeof(float) * LH);
  float*    g32  = (float*)carve(sizeof(float) * LH);  // aliased below as FFN mid (f16, L x 2H)
  float*    fs32 = (float*)carve(sizeof(float) * LH);
  float*    bs32 = (float*)carve(sizeof(float) * LH);
  _Float16* s16  = (_Float16*)carve(sizeof(_Float16) * LH);
  float*    cF   = (float*)carve(sizeof(float) * (size_t)NCHUNK * HDIM);
  float*    cB   = (float*)carve(sizeof(float) * (size_t)NCHUNK * HDIM);
  _Float16* a16  = (_Float16*)g32;  // gate is dead before FFN1 writes this

  // activations: plain f32->f16
  {
    size_t n = (size_t)L_SEQ * 256;
    k_f32_to_f16<<<(int)((n + 2047) / 2048), 256, 0, stream>>>(nf, nf16, (int)n);
  }
  // weights: fused convert + transpose -> [N][K] f16 (once per launch, reused by 64 tiles each)
  dim3 tb(32, 8);
  k_transconv<<<dim3(HDIM / 32, 256 / 32, 1),      tb, 0, stream>>>(Wp, wp16, 256, HDIM);
  k_transconv<<<dim3(HDIM / 32, HDIM / 32, NLAYERS), tb, 0, stream>>>(Wi, wi16, HDIM, HDIM);
  k_transconv<<<dim3(HDIM / 32, HDIM / 32, NLAYERS), tb, 0, stream>>>(Wg, wg16, HDIM, HDIM);
  k_transconv<<<dim3(HDIM / 32, HDIM / 32, NLAYERS), tb, 0, stream>>>(Wo, wo16, HDIM, HDIM);
  k_transconv<<<dim3(2 * HDIM / 32, HDIM / 32, NLAYERS), tb, 0, stream>>>(W1, w116, HDIM, 2 * HDIM);
  k_transconv<<<dim3(HDIM / 32, 2 * HDIM / 32, NLAYERS), tb, 0, stream>>>(W2, w216, 2 * HDIM, HDIM);

#define GEMM(EPI, Ap, Bp, biasp, oF, oH, M_, N_, K_)                                 \
  k_gemm_f16<EPI><<<dim3((N_) / BN, (M_) / BM), 256, 0, stream>>>(Ap, Bp, biasp, oF, \
                                                                  oH, M_, N_, K_)

  // input projection: x = nf @ Wp + bp
  GEMM(0, nf16, wp16, bp, x32, (_Float16*)nullptr, L_SEQ, HDIM, 256);

  const size_t wHH = (size_t)HDIM * HDIM;
  const size_t wH2 = (size_t)HDIM * 2 * HDIM;
  for (int l = 0; l < NLAYERS; ++l) {
    k_layernorm<true><<<L_SEQ, 256, 0, stream>>>(x32, ln1_g + l * HDIM, ln1_b + l * HDIM, h16);
    GEMM(0, h16, wi16 + l * wHH, bi + l * HDIM, u32, (_Float16*)nullptr, L_SEQ, HDIM, HDIM);
    GEMM(1, h16, wg16 + l * wHH, bg + l * HDIM, g32, (_Float16*)nullptr, L_SEQ, HDIM, HDIM);
    k_scan1<<<(HDIM * NCHUNK) / 256, 256, 0, stream>>>(u32, decay + l * HDIM, fs32, bs32, cF, cB);
    k_scan2<<<HDIM / 256, 256, 0, stream>>>(cF, cB, decay + l * HDIM);
    k_scan3<<<(HDIM * NCHUNK) / 256, 256, 0, stream>>>(fs32, bs32, g32, decay + l * HDIM, cF, cB, s16);
    GEMM(3, s16, wo16 + l * wHH, bo + l * HDIM, x32, (_Float16*)nullptr, L_SEQ, HDIM, HDIM);
    k_layernorm<true><<<L_SEQ, 256, 0, stream>>>(x32, ln2_g + l * HDIM, ln2_b + l * HDIM, h16);
    GEMM(2, h16, w116 + l * wH2, b1 + l * 2 * HDIM, (float*)nullptr, a16, L_SEQ, 2 * HDIM, HDIM);
    GEMM(3, a16, w216 + l * wH2, b2 + l * HDIM, x32, (_Float16*)nullptr, L_SEQ, HDIM, 2 * HDIM);
  }
  k_layernorm<false><<<L_SEQ, 256, 0, stream>>>(x32, lnf_g, lnf_b, d_out);
#undef GEMM
}
